// ControGraphConvLayer_12197707120647
// MI455X (gfx1250) — matrix operand order repaired and checked by reading, not compiled, not run on hardware
//
#include <hip/hip_runtime.h>

// ---------------- types ----------------
typedef __attribute__((ext_vector_type(16))) __bf16 v16bf;
typedef __attribute__((ext_vector_type(8)))  __bf16 v8bf;
typedef __attribute__((ext_vector_type(8)))  float  v8f;
typedef __attribute__((ext_vector_type(4)))  float  v4f;

#define N_NODES 8192
#define IN_F    128
#define OUT_F   64

__device__ __forceinline__ v8f wmma_bf16(v16bf a, v16bf b, v8f c) {
    // D = A(16x32 bf16) * B(32x16 bf16) + C(16x16 f32)
    return __builtin_amdgcn_wmma_f32_16x16x32_bf16(
        /*neg_a=*/false, a, /*neg_b=*/false, b,
        /*c_mod=*/(short)0, c, /*reuse_a=*/false, /*reuse_b=*/false);
}

__device__ __forceinline__ v4f nt_ld4(const float* p) {
    return __builtin_nontemporal_load((const v4f*)p);
}

// ---------------- kernel 1: Xw = X @ W^T  (bf16 out) ----------------
__global__ __launch_bounds__(256) void k_xw(const float* __restrict__ X,
                                            const float* __restrict__ W,
                                            __bf16* __restrict__ Xw) {
    int idx = blockIdx.x * 256 + threadIdx.x;   // N_NODES*OUT_F threads
    int m = idx >> 6;        // row of X
    int o = idx & 63;        // output feature
    const float* xr = X + (size_t)m * IN_F;
    const float* wr = W + (size_t)o * IN_F;
    float s = 0.f;
#pragma unroll 8
    for (int k = 0; k < IN_F; ++k) s += xr[k] * wr[k];
    Xw[idx] = (__bf16)s;
}

// ---------------- kernel 2: emb = F_ @ Xw + b ----------------
// Block = 128 threads (4 waves). Block owns 16 rows; each wave owns a K-slice
// of 2048. bf16 WMMA, f32 accumulate. Manual ping-pong double buffering (two
// named register sets, K step 64) -> no inter-set copies; scalar loop bounds
// via readfirstlane -> uniform s_cbranch tail guard.
__global__ __launch_bounds__(128) void k_feat(const float* __restrict__ F,
                                              const __bf16* __restrict__ Xw,
                                              const float* __restrict__ bias,
                                              float* __restrict__ emb) {
    __shared__ float red[4][1024];  // [wave][tile*256 + elem*32 + lane]
    const int tid  = threadIdx.x;
    const int lane = tid & 31;
    const int wave = __builtin_amdgcn_readfirstlane(tid >> 5);  // wave-uniform -> SGPR
    const int m0   = blockIdx.x * 16;
    const int arow = m0 + (lane & 15);
    const int kb   = (lane < 16) ? 0 : 8;            // ISA 16-bit A layout
    const float* frow = F + (size_t)arow * N_NODES + kb;

    v8f c0 = {}, c1 = {}, c2 = {}, c3 = {};
    const int kbeg = wave * 2048;
    const int kend = kbeg + 2048;

    v4f   a0_0, a0_1, a0_2, a0_3, a1_0, a1_1, a1_2, a1_3;
    v16bf b0_0, b0_1, b0_2, b0_3, b1_0, b1_1, b1_2, b1_3;

#define LOADSET(s, kk)                                                  \
    do {                                                                \
        const float* fp_ = frow + (kk);                                 \
        a##s##_0 = nt_ld4(fp_);                                         \
        a##s##_1 = nt_ld4(fp_ + 4);                                     \
        a##s##_2 = nt_ld4(fp_ + 16);                                    \
        a##s##_3 = nt_ld4(fp_ + 20);                                    \
        const __bf16* bp_ = Xw + (size_t)((kk) + kb) * OUT_F + (lane & 15); \
        for (int e_ = 0; e_ < 16; ++e_) {                               \
            size_t o_ = (size_t)(e_ < 8 ? e_ : e_ + 8) * OUT_F;         \
            b##s##_0[e_] = bp_[o_];                                     \
            b##s##_1[e_] = bp_[o_ + 16];                                \
            b##s##_2[e_] = bp_[o_ + 32];                                \
            b##s##_3[e_] = bp_[o_ + 48];                                \
        }                                                               \
    } while (0)

#define COMPUTESET(s)                                                   \
    do {                                                                \
        v16bf a_;                                                       \
        a_[0]  = (__bf16)a##s##_0[0]; a_[1]  = (__bf16)a##s##_0[1];     \
        a_[2]  = (__bf16)a##s##_0[2]; a_[3]  = (__bf16)a##s##_0[3];     \
        a_[4]  = (__bf16)a##s##_1[0]; a_[5]  = (__bf16)a##s##_1[1];     \
        a_[6]  = (__bf16)a##s##_1[2]; a_[7]  = (__bf16)a##s##_1[3];     \
        a_[8]  = (__bf16)a##s##_2[0]; a_[9]  = (__bf16)a##s##_2[1];     \
        a_[10] = (__bf16)a##s##_2[2]; a_[11] = (__bf16)a##s##_2[3];     \
        a_[12] = (__bf16)a##s##_3[0]; a_[13] = (__bf16)a##s##_3[1];     \
        a_[14] = (__bf16)a##s##_3[2]; a_[15] = (__bf16)a##s##_3[3];     \
        c0 = wmma_bf16(a_, b##s##_0, c0);                               \
        c1 = wmma_bf16(a_, b##s##_1, c1);                               \
        c2 = wmma_bf16(a_, b##s##_2, c2);                               \
        c3 = wmma_bf16(a_, b##s##_3, c3);                               \
    } while (0)

    LOADSET(0, kbeg);
#pragma unroll 1
    for (int k0 = kbeg; k0 < kend; k0 += 64) {
        LOADSET(1, k0 + 32);
        __builtin_prefetch(frow + k0 + 768, 0, 0);   // stream F_ ~3KB ahead
        COMPUTESET(0);
        if (k0 + 64 < kend) LOADSET(0, k0 + 64);     // scalar-uniform guard
        COMPUTESET(1);
    }
#undef LOADSET
#undef COMPUTESET

    float* rw = &red[wave][0];
#pragma unroll
    for (int r = 0; r < 8; ++r) {
        rw[0 * 256 + r * 32 + lane] = c0[r];
        rw[1 * 256 + r * 32 + lane] = c1[r];
        rw[2 * 256 + r * 32 + lane] = c2[r];
        rw[3 * 256 + r * 32 + lane] = c3[r];
    }
    __syncthreads();
#pragma unroll
    for (int idx = tid; idx < 1024; idx += 128) {
        float s = red[0][idx] + red[1][idx] + red[2][idx] + red[3][idx];
        int tile = idx >> 8;
        int elem = (idx >> 5) & 7;
        int ln   = idx & 31;
        int row  = m0 + elem + ((ln < 16) ? 0 : 8);   // ISA C-matrix layout
        int col  = tile * 16 + (ln & 15);
        emb[(size_t)row * OUT_F + col] = s + bias[col];
    }
}

// ---------------- kernel 3: sq + bf16 copies (row-major & transposed) ----------------
__global__ __launch_bounds__(256) void k_prep(const float* __restrict__ emb,
                                              float* __restrict__ sq,
                                              __bf16* __restrict__ embB,
                                              __bf16* __restrict__ embT) {
    int r = blockIdx.x * 256 + threadIdx.x;     // N_NODES threads
    const float* e = emb + (size_t)r * OUT_F;
    float s = 0.f;
#pragma unroll
    for (int c = 0; c < OUT_F; ++c) {
        float v = e[c];
        s += v * v;
        __bf16 h = (__bf16)v;
        embB[(size_t)r * OUT_F + c] = h;
        embT[(size_t)c * N_NODES + r] = h;
    }
    sq[r] = s;
}

// ---------------- kernel 4: new_force = dist(emb) * p2p ----------------
// Block = 128 threads (4 waves). Wave computes 16 rows x 64 cols: 8 WMMA
// (G = emb @ emb^T, K=64). All global loads hoisted ahead of the WMMAs so
// they overlap; p2p/force addressed as one base + immediate offsets.
__global__ __launch_bounds__(128) void k_force(const __bf16* __restrict__ embB,
                                               const __bf16* __restrict__ embT,
                                               const float* __restrict__ sq,
                                               const float* __restrict__ p2p,
                                               float* __restrict__ force) {
    const int tid  = threadIdx.x;
    const int lane = tid & 31;
    const int wave = __builtin_amdgcn_readfirstlane(tid >> 5);  // wave-uniform
    const int i0   = blockIdx.x * 16;
    const int j0   = blockIdx.y * 256 + wave * 64;
    const int kb   = (lane < 16) ? 0 : 8;
    const int arow = i0 + (lane & 15);

    // A operand (reused for all 4 column tiles): emb rows i0..i0+15, K=0..63
    const __bf16* ap = embB + (size_t)arow * OUT_F;
    v8bf p0 = *(const v8bf*)(ap + kb);
    v8bf p1 = *(const v8bf*)(ap + 16 + kb);
    v8bf p2 = *(const v8bf*)(ap + 32 + kb);
    v8bf p3 = *(const v8bf*)(ap + 48 + kb);
    v16bf aLo = __builtin_shufflevector(p0, p1, 0,1,2,3,4,5,6,7,8,9,10,11,12,13,14,15);
    v16bf aHi = __builtin_shufflevector(p2, p3, 0,1,2,3,4,5,6,7,8,9,10,11,12,13,14,15);

    // B operands for all 4 tiles, issued up front
    v16bf bt[8];
#pragma unroll
    for (int t = 0; t < 4; ++t) {
        const __bf16* bq = embB + (size_t)(j0 + t * 16 + (lane & 15)) * OUT_F + kb;
        bt[2 * t]     = __builtin_shufflevector(*(const v8bf*)(bq), *(const v8bf*)(bq + 16), 0,1,2,3,4,5,6,7,8,9,10,11,12,13,14,15);
        bt[2 * t + 1] = __builtin_shufflevector(*(const v8bf*)(bq + 32), *(const v8bf*)(bq + 48), 0,1,2,3,4,5,6,7,8,9,10,11,12,13,14,15);
    }

    // p2p tile (streamed once -> non-temporal), single base + imm offsets
    const float* pbase = p2p + (size_t)(i0 + kb) * N_NODES + j0 + (lane & 15);
    float pw[32];
#pragma unroll
    for (int t = 0; t < 4; ++t)
#pragma unroll
        for (int r = 0; r < 8; ++r)
            pw[t * 8 + r] =
                __builtin_nontemporal_load(pbase + (size_t)r * N_NODES + t * 16);

    float sqi[8];
#pragma unroll
    for (int r = 0; r < 8; ++r) sqi[r] = sq[i0 + r + kb];

    v8f c[4];
#pragma unroll
    for (int t = 0; t < 4; ++t) {
        v8f acc = {};
        acc  = wmma_bf16(aLo, bt[2 * t], acc);
        c[t] = wmma_bf16(aHi, bt[2 * t + 1], acc);
    }

    float* fbase = force + (size_t)(i0 + kb) * N_NODES + j0 + (lane & 15);
#pragma unroll
    for (int t = 0; t < 4; ++t) {
        const int col = j0 + t * 16 + (lane & 15);
        const float sqj = sq[col];
#pragma unroll
        for (int r = 0; r < 8; ++r) {
            const int row = i0 + r + kb;
            float d2 = sqi[r] + sqj - 2.0f * c[t][r];
            float out = (d2 > 0.0f && row != col)
                          ? __builtin_sqrtf(d2) * pw[t * 8 + r] : 0.0f;
            __builtin_nontemporal_store(out, fbase + (size_t)r * N_NODES + t * 16);
        }
    }
}

// ---------------- host launcher ----------------
extern "C" void kernel_launch(void* const* d_in, const int* in_sizes, int n_in,
                              void* d_out, int out_size, void* d_ws, size_t ws_size,
                              hipStream_t stream) {
    const float* F_  = (const float*)d_in[0];   // [N,N]
    const float* X   = (const float*)d_in[1];   // [N,128]
    const float* p2p = (const float*)d_in[2];   // [N,N]
    const float* W   = (const float*)d_in[3];   // [64,128]
    const float* b   = (const float*)d_in[4];   // [64]

    float* force = (float*)d_out;                                   // [N,N]
    float* emb   = (float*)d_out + (size_t)N_NODES * N_NODES;       // [N,64]

    char* ws = (char*)d_ws;
    __bf16* Xw   = (__bf16*)(ws);                      // 1 MB
    __bf16* embB = (__bf16*)(ws + (1u << 20));         // 1 MB
    __bf16* embT = (__bf16*)(ws + (2u << 20));         // 1 MB
    float*  sq   = (float*) (ws + (3u << 20));         // 32 KB

    k_xw  <<<(N_NODES * OUT_F) / 256, 256, 0, stream>>>(X, W, Xw);
    k_feat<<<N_NODES / 16, 128, 0, stream>>>(F_, Xw, b, emb);
    k_prep<<<N_NODES / 256, 256, 0, stream>>>(emb, sq, embB, embT);
    dim3 gf(N_NODES / 16, N_NODES / 256);
    k_force<<<gf, 128, 0, stream>>>(embB, embT, sq, p2p, force);
}
